// GGSL_52527450030083
// MI455X (gfx1250) — matrix operand order repaired and checked by reading, not compiled, not run on hardware
//
#include <hip/hip_runtime.h>
#include <math.h>

// ---------------- problem constants (from reference) ----------------
#define NROW   6000
#define NPAD   6016          // round_up(6000,64); = 47*128, multiple of 32
#define F_     512
#define H1_    256
#define H2_    128
#define HID_   256
#define C_     70
#define CPAD_  128
#define KNN_   30
#define HEADS_ 2

typedef __attribute__((ext_vector_type(16))) __bf16 v16bf;
typedef __attribute__((ext_vector_type(8)))  __bf16 v8bf;
typedef __attribute__((ext_vector_type(8)))  float  v8f;

static __device__ __forceinline__ __bf16 f2bf(float f) {
    unsigned u = __builtin_bit_cast(unsigned, f);
    unsigned r = (u + 0x7FFFu + ((u >> 16) & 1u)) >> 16;   // round-nearest-even
    unsigned short s = (unsigned short)r;
    return __builtin_bit_cast(__bf16, s);
}

#define CAT16 0,1,2,3,4,5,6,7,8,9,10,11,12,13,14,15

static __device__ __forceinline__ v16bf wmma_cat(v8bf lo, v8bf hi) {
    return __builtin_shufflevector(lo, hi, CAT16);
}

// ------------------------------------------------------------------
// WMMA bf16 GEMM on zero-padded bf16 operands (no guards in k-loop).
//   C(f32)[M,N] = (addC?C:0) + alpha * [relu] (A @ opB(B))
//   Cbf (optional): padded bf16 copy of the result for the next GEMM.
// Block = 256 threads (8 waves), 128x64 tile; each wave owns a 32x32 tile
// (2x2 grid of 16x16 WMMA accumulators -> 4 WMMA per 32-deep k-step).
// Grid covers padded dims; operand pads hold zeros.
// ------------------------------------------------------------------
__global__ __launch_bounds__(256)
void gemm_bf16_wmma(const __bf16* __restrict__ A, const __bf16* __restrict__ B,
                    float* __restrict__ C, __bf16* __restrict__ Cbf,
                    int M, int N, int K,
                    int lda, int ldb, int ldc, int ldbf,
                    int transB, int addC, int doRelu, float alpha)
{
    __shared__ __bf16 ldsBt[2][64][40];  // [buf][col][k]; stride 40 keeps 16B align

    const int tid  = threadIdx.x;
    const int wave = tid >> 5;
    const int lane = tid & 31;
    const int hs   = lane >> 4;
    const int l16  = lane & 15;

    const int blockM   = blockIdx.y * 128;
    const int blockN   = blockIdx.x * 64;
    const int waveM    = blockM + (wave >> 1) * 32;   // 4 row groups of 32
    const int waveNloc = (wave & 1) * 32;             // 2 col groups of 32
    const int waveN    = blockN + waveNloc;

    const int arow  = waveM + l16;
    const int c0loc = waveNloc + l16;
    const int c1loc = waveNloc + 16 + l16;
    const int ncol0 = blockN + c0loc;
    const int ncol1 = blockN + c1loc;

    // A fragments (contiguous k order, same as B): lane holds K = 16*hs+{0..15}
    const __bf16* Ap0 = A + (long long)arow * lda + 16 * hs;          // rows waveM..+15
    const __bf16* Ap1 = Ap0 + (long long)16 * lda;                    // rows waveM+16..+31

    v8f acc00 = {}, acc01 = {}, acc10 = {}, acc11 = {};

    if (transB) {
        // B[N,K]: lane group hs holds K = 16*hs + 0..15, contiguous in memory
        const __bf16* Bp0 = B + (long long)ncol0 * ldb + 16 * hs;
        const __bf16* Bp1 = B + (long long)ncol1 * ldb + 16 * hs;
        for (int k0 = 0; k0 < K; k0 += 32) {
            v16bf a0 = wmma_cat(*(const v8bf*)(Ap0 + k0), *(const v8bf*)(Ap0 + k0 + 8));
            v16bf a1 = wmma_cat(*(const v8bf*)(Ap1 + k0), *(const v8bf*)(Ap1 + k0 + 8));
            v16bf b0 = wmma_cat(*(const v8bf*)(Bp0 + k0), *(const v8bf*)(Bp0 + k0 + 8));
            v16bf b1 = wmma_cat(*(const v8bf*)(Bp1 + k0), *(const v8bf*)(Bp1 + k0 + 8));
            acc00 = __builtin_amdgcn_wmma_f32_16x16x32_bf16(false, a0, false, b0, (short)0, acc00, false, false);
            acc01 = __builtin_amdgcn_wmma_f32_16x16x32_bf16(false, a0, false, b1, (short)0, acc01, false, false);
            acc10 = __builtin_amdgcn_wmma_f32_16x16x32_bf16(false, a1, false, b0, (short)0, acc10, false, false);
            acc11 = __builtin_amdgcn_wmma_f32_16x16x32_bf16(false, a1, false, b1, (short)0, acc11, false, false);
        }
    } else {
        // Double-buffered LDS staging of transposed 32(K) x 64(N) B tiles.
        const int skk = tid >> 3;        // 0..31  K within tile
        const int sc  = (tid & 7) * 8;   // col group within tile
        {   // prologue: stage tile 0 into buf 0
            v8bf t = *(const v8bf*)(B + (long long)skk * ldb + blockN + sc);
            #pragma unroll
            for (int e = 0; e < 8; ++e) ldsBt[0][sc + e][skk] = t[e];
        }
        int buf = 0;
        for (int k0 = 0; k0 < K; k0 += 32) {
            __syncthreads();             // staged tile for this step is ready
            if (k0 + 32 < K) {           // prefetch next tile into other buffer
                v8bf t = *(const v8bf*)(B + (long long)(k0 + 32 + skk) * ldb + blockN + sc);
                #pragma unroll
                for (int e = 0; e < 8; ++e) ldsBt[buf ^ 1][sc + e][skk] = t[e];
            }
            v16bf a0 = wmma_cat(*(const v8bf*)(Ap0 + k0), *(const v8bf*)(Ap0 + k0 + 8));
            v16bf a1 = wmma_cat(*(const v8bf*)(Ap1 + k0), *(const v8bf*)(Ap1 + k0 + 8));
            v16bf b0 = wmma_cat(*(const v8bf*)(&ldsBt[buf][c0loc][16 * hs]),
                                *(const v8bf*)(&ldsBt[buf][c0loc][16 * hs + 8]));
            v16bf b1 = wmma_cat(*(const v8bf*)(&ldsBt[buf][c1loc][16 * hs]),
                                *(const v8bf*)(&ldsBt[buf][c1loc][16 * hs + 8]));
            acc00 = __builtin_amdgcn_wmma_f32_16x16x32_bf16(false, a0, false, b0, (short)0, acc00, false, false);
            acc01 = __builtin_amdgcn_wmma_f32_16x16x32_bf16(false, a0, false, b1, (short)0, acc01, false, false);
            acc10 = __builtin_amdgcn_wmma_f32_16x16x32_bf16(false, a1, false, b0, (short)0, acc10, false, false);
            acc11 = __builtin_amdgcn_wmma_f32_16x16x32_bf16(false, a1, false, b1, (short)0, acc11, false, false);
            buf ^= 1;
        }
    }

    // ---- epilogue: C/D layout row = r + 8*hs (within 16x16), col = lane&15 ----
    const bool haveBf = (Cbf != nullptr);
    #pragma unroll
    for (int mt = 0; mt < 2; ++mt) {
        v8f aL = mt ? acc10 : acc00;
        v8f aR = mt ? acc11 : acc01;
        #pragma unroll
        for (int r = 0; r < 8; ++r) {
            int row = waveM + mt * 16 + r + 8 * hs;   // < Mpad by grid construction
            float v0 = aL[r], v1 = aR[r];
            if (doRelu) { v0 = fmaxf(v0, 0.0f); v1 = fmaxf(v1, 0.0f); }
            v0 *= alpha; v1 *= alpha;
            bool rOK = row < M;
            if (rOK && ncol0 < N) {
                long long i = (long long)row * ldc + ncol0;
                C[i] = (addC ? C[i] : 0.0f) + v0;
            }
            if (rOK && ncol1 < N) {
                long long i = (long long)row * ldc + ncol1;
                C[i] = (addC ? C[i] : 0.0f) + v1;
            }
            if (haveBf) {  // grid covers whole padded bf16 buffer; pads get 0
                Cbf[(long long)row * ldbf + ncol0] = f2bf((rOK && ncol0 < N) ? v0 : 0.0f);
                Cbf[(long long)row * ldbf + ncol1] = f2bf((rOK && ncol1 < N) ? v1 : 0.0f);
            }
        }
    }
}

// ---------------- f32 -> zero-padded bf16 ----------------
__global__ __launch_bounds__(256)
void convert_pad(const float* __restrict__ src, __bf16* __restrict__ dst,
                 int rows, int cols, int rpad, int cpad)
{
    long long idx = (long long)blockIdx.x * 256 + threadIdx.x;
    if (idx >= (long long)rpad * cpad) return;
    int i = (int)(idx / cpad), j = (int)(idx % cpad);
    float v = (i < rows && j < cols) ? src[(long long)i * cols + j] : 0.0f;
    dst[idx] = f2bf(v);
}

// ---------------- sym_norm helpers ----------------
__global__ __launch_bounds__(256)
void row_deg_rsqrt(const float* __restrict__ A, float* __restrict__ dinv, int n)
{
    __shared__ float red[256];
    int row = blockIdx.x, tid = threadIdx.x;
    float s = 0.0f;
    for (int j = tid; j < n; j += 256) s += A[(long long)row * n + j];
    red[tid] = s; __syncthreads();
    for (int st = 128; st > 0; st >>= 1) {
        if (tid < st) red[tid] += red[tid + st];
        __syncthreads();
    }
    if (tid == 0) {
        float d = red[0];
        dinv[row] = (d > 0.0f) ? rsqrtf(d) : 0.0f;
    }
}

// D^-1/2 A D^-1/2 written directly as zero-padded bf16
__global__ __launch_bounds__(256)
void sym_norm_bf16(const float* __restrict__ A, const float* __restrict__ dinv,
                   __bf16* __restrict__ out, int n, int npad)
{
    long long idx = (long long)blockIdx.x * 256 + threadIdx.x;
    if (idx >= (long long)npad * npad) return;
    int i = (int)(idx / npad), j = (int)(idx % npad);
    float v = (i < n && j < n) ? dinv[i] * dinv[j] * A[(long long)i * n + j] : 0.0f;
    out[idx] = f2bf(v);
}

// ---------------- multi-head weighted-cosine rows (f32 in, padded bf16 out) --
__global__ __launch_bounds__(128)
void metric_eh(const float* __restrict__ emb, const float* __restrict__ wm,
               __bf16* __restrict__ eh, int n, int npad, int d)
{
    int row = blockIdx.x, h = blockIdx.y, t = threadIdx.x;
    long long o = ((long long)h * npad + row) * d + t;
    if (row >= n) { eh[o] = f2bf(0.0f); return; }   // uniform per block
    __shared__ float red[128];
    float v = emb[(long long)row * d + t] * wm[h * d + t];
    red[t] = v * v; __syncthreads();
    for (int st = 64; st > 0; st >>= 1) {
        if (t < st) red[t] += red[t + st];
        __syncthreads();
    }
    float nrm = fmaxf(sqrtf(red[0]), 1e-8f);
    eh[o] = f2bf(v / nrm);
}

// ---------------- per-row top-K (K=30) -> dense sparse row ----------------
__global__ __launch_bounds__(256)
void topk_rows(const float* __restrict__ S, float* __restrict__ Araw, int n)
{
    __shared__ float srow[NROW];
    __shared__ float redv[256];
    __shared__ int   redi[256];
    __shared__ float topv[KNN_];
    __shared__ int   topi[KNN_];

    int row = blockIdx.x, tid = threadIdx.x;
    for (int j = tid; j < n; j += 256) srow[j] = S[(long long)row * n + j];
    __syncthreads();

    for (int sel = 0; sel < KNN_; ++sel) {
        float bv = -INFINITY; int bi = 0x7fffffff;
        for (int j = tid; j < n; j += 256) {
            float v = srow[j];
            if (v > bv || (v == bv && j < bi)) { bv = v; bi = j; }
        }
        redv[tid] = bv; redi[tid] = bi; __syncthreads();
        for (int st = 128; st > 0; st >>= 1) {
            if (tid < st) {
                if (redv[tid + st] > redv[tid] ||
                    (redv[tid + st] == redv[tid] && redi[tid + st] < redi[tid])) {
                    redv[tid] = redv[tid + st]; redi[tid] = redi[tid + st];
                }
            }
            __syncthreads();
        }
        if (tid == 0) {
            topv[sel] = redv[0]; topi[sel] = redi[0];
            srow[redi[0]] = -INFINITY;
        }
        __syncthreads();
    }

    for (int j = tid; j < n; j += 256) {
        float v = 0.0f;
        #pragma unroll
        for (int t = 0; t < KNN_; ++t)
            if (topi[t] == j) v = topv[t];
        Araw[(long long)row * n + j] = v;
    }
}

// ---------------- in-place symmetrize + interpolate ----------------
__global__ __launch_bounds__(256)
void sym_interp(float* __restrict__ Anew, const float* __restrict__ Adj,
                float* __restrict__ Afinal, long long total, int n)
{
    long long idx = (long long)blockIdx.x * 256 + threadIdx.x;
    if (idx >= total) return;
    int i = (int)(idx / n), j = (int)(idx % n);
    if (i > j) return;
    long long ij = (long long)i * n + j, ji = (long long)j * n + i;
    float s = 0.5f * (Anew[ij] + Anew[ji]);
    Anew[ij] = s; Anew[ji] = s;
    Afinal[ij] = s + Adj[ij];
    Afinal[ji] = s + Adj[ji];
}

// ------------------------------------------------------------------
extern "C" void kernel_launch(void* const* d_in, const int* in_sizes, int n_in,
                              void* d_out, int out_size, void* d_ws, size_t ws_size,
                              hipStream_t stream)
{
    const float* x      = (const float*)d_in[0];   // [N,F]
    const float* Adj    = (const float*)d_in[1];   // [N,N]
    const float* Wenc1  = (const float*)d_in[2];   // [F,H1]
    const float* Wenc2  = (const float*)d_in[3];   // [H1,H2]
    const float* wmet   = (const float*)d_in[4];   // [HEADS,H2]
    const float* Wtask1 = (const float*)d_in[5];   // [F,HID]
    const float* Wtask2 = (const float*)d_in[6];   // [HID,C]

    const int n = NROW;
    const long long nn = (long long)n * n;

    // ---- output layout: logits | A_new | A_final | emb ----
    float* logits_o = (float*)d_out;                 // N*C
    float* Anew_o   = logits_o + (long long)n * C_;  // N*N
    float* Afinal_o = Anew_o + nn;                   // N*N
    float* emb_o    = Afinal_o + nn;                 // N*H2

    // ---- workspace layout ----
    float* S    = (float*)d_ws;                      // N*N f32
    float* B1f  = S + nn;                            // N*H1
    float* B2f  = B1f + (long long)n * H1_;          // N*H1
    float* B3f  = B2f + (long long)n * H1_;          // N*H2 (also holds N*C)
    float* dinv = B3f + (long long)n * H2_;          // N
    __bf16* Anbf = (__bf16*)(dinv + n);              // NPAD*NPAD
    __bf16* xbf  = Anbf + (long long)NPAD * NPAD;    // NPAD*F
    __bf16* B1bf = xbf  + (long long)NPAD * F_;      // NPAD*H1
    __bf16* B2bf = B1bf + (long long)NPAD * H1_;     // NPAD*H1
    __bf16* B3bf = B2bf + (long long)NPAD * H1_;     // NPAD*CPAD (>= NPAD*H2)
    __bf16* EHbf = B3bf + (long long)NPAD * CPAD_;   // HEADS*NPAD*H2
    __bf16* Wabf = EHbf + (long long)HEADS_ * NPAD * H2_; // F*H1
    __bf16* Wbbf = Wabf + (long long)F_ * H1_;       // H1*CPAD

    dim3 blk(256);
    auto cblocks = [](long long total) { return (unsigned)((total + 255) / 256); };

    auto gemm = [&](const __bf16* A, const __bf16* B, float* C, __bf16* Cbf,
                    int M, int N, int K, int lda, int ldb, int ldc, int ldbf,
                    int transB, int addC, int relu, float alpha) {
        int ngrid = Cbf ? (ldbf / 64) : ((N + 63) / 64);
        dim3 grid((unsigned)ngrid, (unsigned)((M + 127) / 128));
        gemm_bf16_wmma<<<grid, blk, 0, stream>>>(A, B, C, Cbf, M, N, K,
                                                 lda, ldb, ldc, ldbf,
                                                 transB, addC, relu, alpha);
    };

    // 0) convert inputs to padded bf16
    convert_pad<<<cblocks((long long)NPAD * F_), blk, 0, stream>>>(x, xbf, n, F_, NPAD, F_);
    convert_pad<<<cblocks((long long)F_ * H1_), blk, 0, stream>>>(Wenc1, Wabf, F_, H1_, F_, H1_);
    convert_pad<<<cblocks((long long)H1_ * H2_), blk, 0, stream>>>(Wenc2, Wbbf, H1_, H2_, H1_, H2_);

    // 1) An = sym_norm(Adj) directly as padded bf16
    row_deg_rsqrt<<<n, blk, 0, stream>>>(Adj, dinv, n);
    sym_norm_bf16<<<cblocks((long long)NPAD * NPAD), blk, 0, stream>>>(Adj, dinv, Anbf, n, NPAD);

    // 2) encoder: emb = An @ ( relu(An @ (x@Wenc1)) @ Wenc2 )
    gemm(xbf,  Wabf, B1f,   B1bf,    n, H1_, F_,   F_,   H1_, H1_, H1_, 0, 0, 0, 1.0f);
    gemm(Anbf, B1bf, B2f,   B2bf,    n, H1_, NPAD, NPAD, H1_, H1_, H1_, 0, 0, 1, 1.0f);
    gemm(B2bf, Wbbf, B3f,   B3bf,    n, H2_, H1_,  H1_,  H2_, H2_, H2_, 0, 0, 0, 1.0f);
    gemm(Anbf, B3bf, emb_o, nullptr, n, H2_, NPAD, NPAD, H2_, H2_, 0,   0, 0, 0, 1.0f);

    // 3) S = (1/HEADS) * sum_h eh_h @ eh_h^T
    metric_eh<<<dim3(NPAD, HEADS_), dim3(128), 0, stream>>>(emb_o, wmet, EHbf, n, NPAD, H2_);
    gemm(EHbf, EHbf, S, nullptr, n, n, H2_, H2_, H2_, n, 0, 1, 0, 0, 1.0f / HEADS_);
    gemm(EHbf + (long long)NPAD * H2_, EHbf + (long long)NPAD * H2_,
         S, nullptr, n, n, H2_, H2_, H2_, n, 0, 1, 1, 0, 1.0f / HEADS_);

    // 4) per-row top-K -> raw A_new
    topk_rows<<<n, blk, 0, stream>>>(S, Anew_o, n);

    // 5) symmetrize in place + interpolate with Adj
    sym_interp<<<cblocks(nn), blk, 0, stream>>>(Anew_o, Adj, Afinal_o, nn, n);

    // 6) Anf = sym_norm(A_final) (reuse Anbf)
    row_deg_rsqrt<<<n, blk, 0, stream>>>(Afinal_o, dinv, n);
    sym_norm_bf16<<<cblocks((long long)NPAD * NPAD), blk, 0, stream>>>(Afinal_o, dinv, Anbf, n, NPAD);

    // 7) task GCN: logits = Anf @ ( relu(Anf @ (x@Wtask1)) @ Wtask2 )
    convert_pad<<<cblocks((long long)F_ * HID_), blk, 0, stream>>>(Wtask1, Wabf, F_, HID_, F_, HID_);
    convert_pad<<<cblocks((long long)HID_ * CPAD_), blk, 0, stream>>>(Wtask2, Wbbf, HID_, C_, HID_, CPAD_);
    gemm(xbf,  Wabf, B1f,      B1bf,    n, HID_, F_,   F_,   HID_,  HID_, HID_,  0, 0, 0, 1.0f);
    gemm(Anbf, B1bf, B2f,      B2bf,    n, HID_, NPAD, NPAD, HID_,  HID_, HID_,  0, 0, 1, 1.0f);
    gemm(B2bf, Wbbf, B3f,      B3bf,    n, C_,   HID_, HID_, CPAD_, C_,   CPAD_, 0, 0, 0, 1.0f);
    gemm(Anbf, B3bf, logits_o, nullptr, n, C_,   NPAD, NPAD, CPAD_, C_,   0,     0, 0, 0, 1.0f);
}
